// EnMutiAttenNet_2113123909957
// MI455X (gfx1250) — matrix operand order repaired and checked by reading, not compiled, not run on hardware
//
#include <hip/hip_runtime.h>
#include <math.h>

// ---------------------------------------------------------------------------
// LCCNet / RAFT forward for MI455X (gfx1250, wave32, WMMA).
// All convs + correlation run as f16 WMMA implicit GEMM (f32 accumulate).
// Activations: NHWC f16 (K-dim contiguous). Weights: device-packed into the
// V_WMMA_F32_16X16X32_F16 B-operand lane layout -> one 32B load per lane/step.
// A-gather index math is amortized: one k->(dy,dx,ci) decomposition per
// 8-element contiguous run + increment/carry, instead of per element.
// ---------------------------------------------------------------------------

typedef __attribute__((ext_vector_type(16))) _Float16 v16h;
typedef __attribute__((ext_vector_type(8)))  _Float16 v8h;
typedef __attribute__((ext_vector_type(8)))  float    v8f;

#define NB   8
#define HIN  256
#define WIN  512
#define HF   16
#define WF   32
#define NPIX (NB*HF*WF)          // 4096 coarse pixels
#define CORR_CH 324              // 4 levels * 81

#define ACT_NONE 0
#define ACT_RELU 1
#define ACT_SIG  2
#define ACT_TANH 3

__device__ __forceinline__ float act_apply(float v, int act) {
  if (act == ACT_RELU) return fmaxf(v, 0.f);
  if (act == ACT_SIG)  return 1.f / (1.f + __expf(-v));
  if (act == ACT_TANH) return tanhf(v);
  return v;
}

// ---------------------------------------------------------------------------
// Weight packing: w[O,I,kh,kw] f32 -> wmma-ready f16 tiles [Ntile][Ktile][32][16]
// value(ntile,ktile,lane,i) = w[n][ci][dy][dx],  n = ntile*16 + (lane&15),
// k = ktile*32 + (lane>>4)*8 + (i<8 ? i : i+8), k -> (dy,dx,ci) with ci fastest (NHWC im2col).
// Zero-padded beyond O / K so WMMA tiles are always safe to stream.
// ---------------------------------------------------------------------------
__global__ void pack_conv_w_kernel(const float* __restrict__ w, _Float16* __restrict__ wp,
                                   int O, int I, int kh, int kw, int total) {
  int idx = blockIdx.x * blockDim.x + threadIdx.x;
  if (idx >= total) return;
  int K = kh * kw * I;
  int Ktiles = (K + 31) >> 5;
  int i     = idx & 15;
  int lane  = (idx >> 4) & 31;
  int tile  = idx >> 9;
  int ktile = tile % Ktiles;
  int ntile = tile / Ktiles;
  int n = ntile * 16 + (lane & 15);
  int k = ktile * 32 + (lane >> 4) * 8 + (i < 8 ? i : i + 8);
  float v = 0.f;
  if (n < O && k < K) {
    int dy = k / (kw * I); int r = k - dy * (kw * I);
    int dx = r / I;        int ci = r - dx * I;
    v = w[(((size_t)n * I + ci) * kh + dy) * kw + dx];
  }
  wp[idx] = (_Float16)v;
}

// ---------------------------------------------------------------------------
// Implicit-GEMM conv, one wave computes one 16(M pixels) x 16(N out-ch) tile.
// A fragment (16x32 f16): lane layout per CDNA5 ISA 7.12.2 -- lane holds row
// m = lane&15, K chunk (lane>>4)*8 + {i<8 ? i : i+8}. B fragment: single
// 32-byte vector load from the packed weight tiles (prefetched one K-step
// ahead via global_prefetch). blockDim = (32,4): 4 waves share one B stream.
// ---------------------------------------------------------------------------
__device__ __forceinline__ void kdecomp(int k, int kwC, int Cin,
                                        int& dy, int& dx, int& ci) {
  dy = k / kwC; int r2 = k - dy * kwC;
  dx = r2 / Cin; ci = r2 - dx * Cin;
}

__global__ __launch_bounds__(128)
void conv_wmma_kernel(const _Float16* __restrict__ in,
                      const _Float16* __restrict__ wpack,
                      const float* __restrict__ bias,
                      _Float16* __restrict__ out,
                      int Nimg, int Hin, int Win, int Cin, int CinStride,
                      int Hout, int Wout, int Cout, int CoutStride, int coutOff,
                      int kh, int kw, int stride, int act) {
  const int lane = threadIdx.x;
  const int l15  = lane & 15;
  const int h16  = lane >> 4;
  const int mtile = blockIdx.x * 4 + threadIdx.y;
  const int ntile = blockIdx.y;
  const int M = Nimg * Hout * Wout;
  const int K = kh * kw * Cin;
  const int kwC = kw * Cin;
  const int Ktiles = (K + 31) >> 5;

  int m = mtile * 16 + l15;
  const bool mvalid = (m < M);
  int mm = mvalid ? m : 0;
  int nimg = mm / (Hout * Wout);
  int rem  = mm - nimg * (Hout * Wout);
  int oy = rem / Wout;
  int ox = rem - oy * Wout;
  const int ph = kh >> 1, pw = kw >> 1;
  const int iy0 = oy * stride - ph;
  const int ix0 = ox * stride - pw;
  const _Float16* rowbase = in + (size_t)nimg * Hin * Win * CinStride;
  const _Float16* wt = wpack + ((size_t)ntile * Ktiles) * 512 + lane * 16;
  const bool onebyone = (kh == 1) && (kw == 1) && (CinStride == Cin) && ((Cin & 7) == 0);

  v8f acc = {};
  #pragma unroll 2
  for (int kt = 0; kt < Ktiles; ++kt) {
    // prefetch next packed B tile into cache while this step computes
    if (kt + 1 < Ktiles) __builtin_prefetch(wt + (size_t)(kt + 1) * 512, 0, 0);
    v16h b = *(const v16h*)(wt + (size_t)kt * 512);
    v16h a;
    const int kbase = kt * 32 + h16 * 8;
    if (onebyone && mvalid && (kbase + 24) <= Cin) {
      // 1x1 conv fast path: both 8-element K runs are contiguous & 16B aligned
      const _Float16* src = in + (size_t)mm * Cin + kbase;
      v8h lo = *(const v8h*)(src);
      v8h hi = *(const v8h*)(src + 16);
      a = __builtin_shufflevector(lo, hi, 0,1,2,3,4,5,6,7,8,9,10,11,12,13,14,15);
    } else {
      // two contiguous 8-element runs: decompose once per run, then
      // increment (ci,dx,dy) with carries -- no per-element div/mod.
      #pragma unroll
      for (int half = 0; half < 2; ++half) {
        int k = kbase + half * 16;
        int dy, dx, ci;
        kdecomp(k, kwC, Cin, dy, dx, ci);
        #pragma unroll
        for (int i = 0; i < 8; ++i) {
          float v = 0.f;
          int iy = iy0 + dy, ix = ix0 + dx;
          if (mvalid && k < K && iy >= 0 && iy < Hin && ix >= 0 && ix < Win)
            v = (float)rowbase[((size_t)iy * Win + ix) * CinStride + ci];
          a[half * 8 + i] = (_Float16)v;
          ++k; ++ci;
          if (ci == Cin) { ci = 0; ++dx; if (dx == kw) { dx = 0; ++dy; } }
        }
      }
    }
    acc = __builtin_amdgcn_wmma_f32_16x16x32_f16(false, a, false, b, (short)0, acc,
                                                 false, false);
  }
  // D layout: vgpr r -> M = r + 8*(lane>=16), N = lane&15
  int nloc = ntile * 16 + l15;
  #pragma unroll
  for (int r = 0; r < 8; ++r) {
    int mrow = mtile * 16 + r + h16 * 8;
    if (mrow < M && nloc < Cout) {
      float v = acc[r] + bias[nloc];
      v = act_apply(v, act);
      out[(size_t)mrow * CoutStride + coutOff + nloc] = (_Float16)v;
    }
  }
}

// ---------------------------------------------------------------------------
// Correlation: per image, corr = F1[512,384] x F2^T[384,512] / sqrt(384).
// Both fragments are fully contiguous vector loads (NHWC, K = channel dim).
// ---------------------------------------------------------------------------
__global__ __launch_bounds__(32)
void corr_wmma_kernel(const _Float16* __restrict__ f1, const _Float16* __restrict__ f2,
                      _Float16* __restrict__ out, float scale) {
  const int lane = threadIdx.x;
  const int l15 = lane & 15, h16 = lane >> 4;
  const int n = blockIdx.z;
  const _Float16* A = f1 + (size_t)n * 512 * 384;
  const _Float16* B = f2 + (size_t)n * 512 * 384;
  const int mrow = blockIdx.x * 16 + l15;
  const int ncol = blockIdx.y * 16 + l15;
  v8f acc = {};
  #pragma unroll
  for (int k0 = 0; k0 < 384; k0 += 32) {
    const _Float16* ap = A + (size_t)mrow * 384 + k0 + h16 * 16;
    v8h lo = *(const v8h*)ap;
    v8h hi = *(const v8h*)(ap + 8);
    v16h a = __builtin_shufflevector(lo, hi, 0,1,2,3,4,5,6,7,8,9,10,11,12,13,14,15);
    v16h b = *(const v16h*)(B + (size_t)ncol * 384 + k0 + h16 * 16);
    acc = __builtin_amdgcn_wmma_f32_16x16x32_f16(false, a, false, b, (short)0, acc,
                                                 false, false);
  }
  #pragma unroll
  for (int r = 0; r < 8; ++r) {
    int mr = blockIdx.x * 16 + r + h16 * 8;
    out[((size_t)n * 512 + mr) * 512 + blockIdx.y * 16 + l15] = (_Float16)(acc[r] * scale);
  }
}

// ---------------------------------------------------------------------------
// Instance norm + ReLU, one block per (image, channel), NHWC in-place.
// ---------------------------------------------------------------------------
__global__ __launch_bounds__(256)
void inorm_relu_kernel(_Float16* __restrict__ buf, int npix, int C) {
  const int n = blockIdx.x / C, c = blockIdx.x % C;
  _Float16* base = buf + ((size_t)n * npix) * C + c;
  float s = 0.f, s2 = 0.f;
  for (int i = threadIdx.x; i < npix; i += blockDim.x) {
    float v = (float)base[(size_t)i * C];
    s += v; s2 += v * v;
  }
  __shared__ float rs[256], rq[256];
  rs[threadIdx.x] = s; rq[threadIdx.x] = s2;
  __syncthreads();
  for (int st = 128; st > 0; st >>= 1) {
    if (threadIdx.x < st) { rs[threadIdx.x] += rs[threadIdx.x + st];
                            rq[threadIdx.x] += rq[threadIdx.x + st]; }
    __syncthreads();
  }
  float mean = rs[0] / (float)npix;
  float var  = rq[0] / (float)npix - mean * mean;
  float inv  = rsqrtf(fmaxf(var, 0.f) + 1e-5f);
  for (int i = threadIdx.x; i < npix; i += blockDim.x) {
    float v = ((float)base[(size_t)i * C] - mean) * inv;
    base[(size_t)i * C] = (_Float16)fmaxf(v, 0.f);
  }
}

// ------------------------- small elementwise kernels ------------------------
__global__ void prep_rgb_kernel(const float* __restrict__ src, _Float16* __restrict__ dst) {
  int idx = blockIdx.x * blockDim.x + threadIdx.x;
  const int total = NB * HIN * WIN * 3;
  if (idx >= total) return;
  int c = idx % 3; int pix = idx / 3;
  int n = pix / (HIN * WIN); int r = pix - n * HIN * WIN;
  int y = r / WIN; int x = r - y * WIN;
  float v = src[(((size_t)n * 3 + c) * HIN + y) * WIN + x];
  dst[idx] = (_Float16)(2.f * (v / 255.f) - 1.f);
}

__global__ void prep_dep_kernel(const float* __restrict__ src, _Float16* __restrict__ dst) {
  int idx = blockIdx.x * blockDim.x + threadIdx.x;
  const int total = NB * HIN * WIN;
  if (idx >= total) return;
  dst[idx] = (_Float16)(2.f * src[idx] - 1.f);
}

__global__ void pool2_kernel(const _Float16* __restrict__ src, _Float16* __restrict__ dst,
                             int ih, int iw, int total) {
  int idx = blockIdx.x * blockDim.x + threadIdx.x;
  if (idx >= total) return;
  int oh = ih >> 1, ow = iw >> 1;
  int b = idx / (oh * ow); int r = idx - b * (oh * ow);
  int oy = r / ow, ox = r - oy * ow;
  const _Float16* s = src + (size_t)b * ih * iw;
  float v = ((float)s[(2*oy)*iw + 2*ox]   + (float)s[(2*oy)*iw + 2*ox+1] +
             (float)s[(2*oy+1)*iw + 2*ox] + (float)s[(2*oy+1)*iw + 2*ox+1]) * 0.25f;
  dst[idx] = (_Float16)v;
}

__global__ void corr_lookup_kernel(const _Float16* __restrict__ l0, const _Float16* __restrict__ l1,
                                   const _Float16* __restrict__ l2, const _Float16* __restrict__ l3,
                                   const float* __restrict__ coords, _Float16* __restrict__ out) {
  int idx = blockIdx.x * blockDim.x + threadIdx.x;
  const int total = NPIX * CORR_CH;
  if (idx >= total) return;
  int c = idx % CORR_CH; int pix = idx / CORR_CH;
  int n = pix >> 9; int r = pix & 511;
  float cx = coords[((size_t)n * 2 + 0) * 512 + r];
  float cy = coords[((size_t)n * 2 + 1) * 512 + r];
  int lvl = c / 81; int rr = c - lvl * 81;
  int dyi = rr / 9, dxi = rr - dyi * 9;
  float px = cx / (float)(1 << lvl) + (float)(dxi - 4);
  float py = cy / (float)(1 << lvl) + (float)(dyi - 4);
  int hl = HF >> lvl, wl = WF >> lvl;
  const _Float16* img = (lvl == 0 ? l0 : lvl == 1 ? l1 : lvl == 2 ? l2 : l3)
                        + (size_t)pix * hl * wl;
  float x0 = floorf(px), y0 = floorf(py);
  float lx = px - x0, ly = py - y0;
  float accv = 0.f;
  #pragma unroll
  for (int j = 0; j < 4; ++j) {
    float xi = x0 + (float)(j & 1), yi = y0 + (float)(j >> 1);
    float wgt = ((j & 1) ? lx : 1.f - lx) * ((j >> 1) ? ly : 1.f - ly);
    if (xi >= 0.f && xi < (float)wl && yi >= 0.f && yi < (float)hl)
      accv += wgt * (float)img[(int)yi * wl + (int)xi];
  }
  out[idx] = (_Float16)accv;
}

__global__ void init_coords_kernel(float* __restrict__ coords) {
  int idx = blockIdx.x * blockDim.x + threadIdx.x;
  if (idx >= NB * 2 * HF * WF) return;
  int x = idx & 31; int ch = (idx >> 9) & 1;
  int y = (idx >> 5) & 15;
  coords[idx] = (ch == 0) ? (float)x : (float)y;
}

__global__ void flow_kernel(const float* __restrict__ coords, _Float16* __restrict__ xbuf) {
  int idx = blockIdx.x * blockDim.x + threadIdx.x;
  if (idx >= NPIX) return;
  int n = idx >> 9; int r = idx & 511;
  int y = r >> 5, x = r & 31;
  float cx = coords[((size_t)n * 2 + 0) * 512 + r];
  float cy = coords[((size_t)n * 2 + 1) * 512 + r];
  xbuf[(size_t)idx * 320 + 318] = (_Float16)(cx - (float)x);
  xbuf[(size_t)idx * 320 + 319] = (_Float16)(cy - (float)y);
}

__global__ void copy_ch_kernel(const _Float16* __restrict__ src, int sStride, int sOff,
                               _Float16* __restrict__ dst, int dStride, int dOff,
                               int nch, int total) {
  int idx = blockIdx.x * blockDim.x + threadIdx.x;
  if (idx >= total) return;
  int p = idx / nch; int c = idx - p * nch;
  dst[(size_t)p * dStride + dOff + c] = src[(size_t)p * sStride + sOff + c];
}

__global__ void init_net_inp_kernel(const _Float16* __restrict__ cn,
                                    _Float16* __restrict__ net, _Float16* __restrict__ xbuf) {
  int idx = blockIdx.x * blockDim.x + threadIdx.x;
  if (idx >= NPIX * 192) return;
  int p = idx / 192; int c = idx - p * 192;
  net[idx] = (_Float16)tanhf((float)cn[(size_t)p * 384 + c]);
  xbuf[(size_t)p * 320 + c] = (_Float16)fmaxf((float)cn[(size_t)p * 384 + 192 + c], 0.f);
}

__global__ void rnet_kernel(const _Float16* __restrict__ net, const _Float16* __restrict__ r,
                            _Float16* __restrict__ qin, int total) {
  int idx = blockIdx.x * blockDim.x + threadIdx.x;
  if (idx >= total) return;
  int p = idx / 192; int c = idx - p * 192;
  qin[(size_t)p * 512 + c] = (_Float16)((float)r[idx] * (float)net[idx]);
}

__global__ void gru_update_kernel(_Float16* __restrict__ net, const _Float16* __restrict__ z,
                                  const _Float16* __restrict__ q, int total) {
  int idx = blockIdx.x * blockDim.x + threadIdx.x;
  if (idx >= total) return;
  float zv = (float)z[idx];
  net[idx] = (_Float16)((1.f - zv) * (float)net[idx] + zv * (float)q[idx]);
}

__global__ void add_dflow_kernel(float* __restrict__ coords, const _Float16* __restrict__ dflow) {
  int idx = blockIdx.x * blockDim.x + threadIdx.x;
  if (idx >= NB * 2 * HF * WF) return;
  int x = idx & 31; int y = (idx >> 5) & 15;
  int ch = (idx >> 9) & 1; int n = idx >> 10;
  coords[idx] += (float)dflow[((size_t)((n << 9) | (y * 32 + x))) * 2 + ch];
}

// ---------------------------------------------------------------------------
// FC head: x = coords1 flat [8,1024] -> fc1(512) leaky -> {trasl,rot}(128)
// leaky -> t(3), r(4)/||r||. One block per batch image; 4M MACs -> VALU OK.
// ---------------------------------------------------------------------------
__global__ __launch_bounds__(256)
void fc_head_kernel(const float* __restrict__ coords,
                    const float* __restrict__ w1,  const float* __restrict__ b1,
                    const float* __restrict__ wt1, const float* __restrict__ bt1,
                    const float* __restrict__ wr1, const float* __restrict__ br1,
                    const float* __restrict__ wt2, const float* __restrict__ bt2,
                    const float* __restrict__ wr2, const float* __restrict__ br2,
                    float* __restrict__ out) {
  __shared__ float xin[1024];
  __shared__ float h1[512];
  __shared__ float hb[256];   // [0:128]=trasl branch, [128:256]=rot branch
  __shared__ float rtmp[4];
  const int n = blockIdx.x, tid = threadIdx.x;
  for (int i = tid; i < 1024; i += 256) xin[i] = coords[(size_t)n * 1024 + i];
  __syncthreads();
  for (int o = tid; o < 512; o += 256) {
    const float* wrow = w1 + (size_t)o * 1024;
    float s = b1[o];
    for (int k = 0; k < 1024; ++k) s += wrow[k] * xin[k];
    h1[o] = s > 0.f ? s : 0.1f * s;
  }
  __syncthreads();
  {
    int o = tid & 127;
    const float* wm = (tid < 128) ? wt1 : wr1;
    const float* bm = (tid < 128) ? bt1 : br1;
    const float* wrow = wm + (size_t)o * 512;
    float s = bm[o];
    for (int k = 0; k < 512; ++k) s += wrow[k] * h1[k];
    hb[tid] = s > 0.f ? s : 0.1f * s;
  }
  __syncthreads();
  if (tid < 3) {
    const float* wrow = wt2 + (size_t)tid * 128;
    float s = bt2[tid];
    for (int k = 0; k < 128; ++k) s += wrow[k] * hb[k];
    out[(size_t)n * 3 + tid] = s;                       // t -> out[0:24]
  }
  if (tid >= 32 && tid < 36) {
    int o = tid - 32;
    const float* wrow = wr2 + (size_t)o * 128;
    float s = br2[o];
    for (int k = 0; k < 128; ++k) s += wrow[k] * hb[128 + k];
    rtmp[o] = s;
  }
  __syncthreads();
  if (tid < 4) {
    float nn = sqrtf(rtmp[0]*rtmp[0] + rtmp[1]*rtmp[1] + rtmp[2]*rtmp[2] + rtmp[3]*rtmp[3]);
    nn = fmaxf(nn, 1e-12f);
    out[24 + (size_t)n * 4 + tid] = rtmp[tid] / nn;     // r -> out[24:56]
  }
}

// ---------------------------------------------------------------------------
// Host orchestration
// ---------------------------------------------------------------------------
struct ConvDesc { int O, I, kh, kw, bi, wi; };

// d_in layout: [0]=rgb, [1]=depth, then params flattened as a JAX pytree
// (dicts iterate in sorted key order, each conv dict = {b, w}).
static const ConvDesc g_convs[28] = {
  // cnet (idx 0-4)
  { 64,   3, 7, 7,  2,  3}, { 96,  64, 3, 3,  4,  5}, {128,  96, 3, 3,  6,  7},
  {192, 128, 3, 3,  8,  9}, {384, 192, 1, 1, 10, 11},
  // f_lidar (5-9)
  { 64,   1, 7, 7, 12, 13}, { 96,  64, 3, 3, 14, 15}, {128,  96, 3, 3, 16, 17},
  {192, 128, 3, 3, 18, 19}, {384, 192, 1, 1, 20, 21},
  // f_rgb (10-14)
  { 64,   3, 7, 7, 22, 23}, { 96,  64, 3, 3, 24, 25}, {128,  96, 3, 3, 26, 27},
  {192, 128, 3, 3, 28, 29}, {384, 192, 1, 1, 30, 31},
  // update (sorted): conv, convc1, convc2, convf1, convf2, convq1, convq2,
  //                  convr1, convr2, convz1, convz2, fh1, fh2   (15-27)
  {126, 256, 3, 3, 42, 43}, {256, 324, 1, 1, 44, 45}, {192, 256, 3, 3, 46, 47},
  {128,   2, 7, 7, 48, 49}, { 64, 128, 3, 3, 50, 51},
  {192, 512, 1, 5, 52, 53}, {192, 512, 5, 1, 54, 55},
  {192, 512, 1, 5, 56, 57}, {192, 512, 5, 1, 58, 59},
  {192, 512, 1, 5, 60, 61}, {192, 512, 5, 1, 62, 63},
  {256, 192, 3, 3, 64, 65}, {  2, 256, 3, 3, 66, 67},
};

static inline size_t packHalfs(const ConvDesc& d) {
  int K = d.kh * d.kw * d.I;
  size_t nt = (size_t)(d.O + 15) / 16, kt = (size_t)(K + 31) / 32;
  return nt * kt * 512;
}

extern "C" void kernel_launch(void* const* d_in, const int* in_sizes, int n_in,
                              void* d_out, int out_size, void* d_ws, size_t ws_size,
                              hipStream_t stream) {
  (void)in_sizes; (void)out_size; (void)ws_size;
  if (n_in < 68) return;
  char* ws = (char*)d_ws;
  size_t off = 0;
  auto alloc = [&](size_t bytes) -> size_t {
    off = (off + 255) & ~(size_t)255;
    size_t o = off; off += bytes; return o;
  };

  size_t wpOff[28];
  for (int i = 0; i < 28; ++i) wpOff[i] = alloc(packHalfs(g_convs[i]) * sizeof(_Float16));

  const size_t actA = alloc((size_t)NB * 64 * 128 * 96 * sizeof(_Float16));   // 12.6 MB
  const size_t actB = alloc((size_t)NB * 128 * 256 * 64 * sizeof(_Float16));  // 33.6 MB
  const size_t f1o  = alloc((size_t)NPIX * 384 * sizeof(_Float16));
  const size_t f2o  = alloc((size_t)NPIX * 384 * sizeof(_Float16));
  const size_t cno  = alloc((size_t)NPIX * 384 * sizeof(_Float16));
  const size_t p0o  = alloc((size_t)NB * 512 * 512 * sizeof(_Float16));
  const size_t p1o  = alloc((size_t)NB * 512 * 128 * sizeof(_Float16));
  const size_t p2o  = alloc((size_t)NB * 512 *  32 * sizeof(_Float16));
  const size_t p3o  = alloc((size_t)NB * 512 *   8 * sizeof(_Float16));
  const size_t coordsOff = alloc((size_t)NB * 2 * HF * WF * sizeof(float));

  // Per-iteration buffers alias actB (dead after the encoders; 27.3MB < 33.6MB)
  size_t io = actB;
  auto carve = [&](size_t halfs) -> _Float16* {
    _Float16* p = (_Float16*)(ws + io); io += halfs * sizeof(_Float16); return p;
  };
  _Float16* corrbuf = carve((size_t)NPIX * 324);
  _Float16* corbuf  = carve((size_t)NPIX * 256);
  _Float16* motin   = carve((size_t)NPIX * 256);
  _Float16* flobuf  = carve((size_t)NPIX * 128);
  _Float16* xbuf    = carve((size_t)NPIX * 320); // 0:192 inp | 192:318 mot | 318:320 flow
  _Float16* hxbuf   = carve((size_t)NPIX * 512); // 0:192 net | 192:512 x
  _Float16* qinbuf  = carve((size_t)NPIX * 512); // 0:192 r*net | 192:512 x
  _Float16* zbuf    = carve((size_t)NPIX * 192);
  _Float16* rbuf    = carve((size_t)NPIX * 192);
  _Float16* qbuf    = carve((size_t)NPIX * 192);
  _Float16* netbuf  = carve((size_t)NPIX * 192);
  _Float16* fhbuf   = carve((size_t)NPIX * 256);
  _Float16* dflow   = carve((size_t)NPIX * 2);

  _Float16* f1 = (_Float16*)(ws + f1o);
  _Float16* f2 = (_Float16*)(ws + f2o);
  _Float16* cn = (_Float16*)(ws + cno);
  _Float16* py0 = (_Float16*)(ws + p0o);
  _Float16* py1 = (_Float16*)(ws + p1o);
  _Float16* py2 = (_Float16*)(ws + p2o);
  _Float16* py3 = (_Float16*)(ws + p3o);
  float* coords = (float*)(ws + coordsOff);
  _Float16* inA = (_Float16*)(ws + actA);
  _Float16* pB  = (_Float16*)(ws + actB);

  // ---- pack all conv weights into WMMA B-tile layout (every call; cheap) ----
  for (int i = 0; i < 28; ++i) {
    const ConvDesc& d = g_convs[i];
    int total = (int)packHalfs(d);
    pack_conv_w_kernel<<<(total + 255) / 256, 256, 0, stream>>>(
        (const float*)d_in[d.wi], (_Float16*)(ws + wpOff[i]), d.O, d.I, d.kh, d.kw, total);
  }

  auto conv = [&](const _Float16* in, int Hi, int Wi, int Cin, int CinStride,
                  int ci, _Float16* out2, int Ho, int Wo,
                  int CoutStride, int coutOff, int stride, int act) {
    const ConvDesc& d = g_convs[ci];
    int M = NB * Ho * Wo;
    dim3 grid(((M + 15) / 16 + 3) / 4, (d.O + 15) / 16);
    dim3 block(32, 4);
    conv_wmma_kernel<<<grid, block, 0, stream>>>(
        in, (const _Float16*)(ws + wpOff[ci]), (const float*)d_in[d.bi], out2,
        NB, Hi, Wi, Cin, CinStride, Ho, Wo, d.O, CoutStride, coutOff,
        d.kh, d.kw, stride, act);
  };

  auto encoder = [&](int base, _Float16* dst) {
    struct D { int H, W, C; };
    const D dims[6] = {{256, 512, g_convs[base].I}, {128, 256, 64}, {64, 128, 96},
                       {32, 64, 128}, {16, 32, 192}, {16, 32, 384}};
    _Float16* cur = inA;
    _Float16* nxt = pB;
    for (int s = 0; s < 4; ++s) {
      conv(cur, dims[s].H, dims[s].W, dims[s].C, dims[s].C, base + s, nxt,
           dims[s + 1].H, dims[s + 1].W, dims[s + 1].C, 0, 2, ACT_NONE);
      inorm_relu_kernel<<<NB * dims[s + 1].C, 256, 0, stream>>>(
          nxt, dims[s + 1].H * dims[s + 1].W, dims[s + 1].C);
      _Float16* t = cur; cur = nxt; nxt = t;
    }
    conv(cur, 16, 32, 192, 192, base + 4, dst, 16, 32, 384, 0, 1, ACT_NONE);
  };

  // ---- encoders -----------------------------------------------------------
  {
    int t = NB * HIN * WIN * 3;
    prep_rgb_kernel<<<(t + 255) / 256, 256, 0, stream>>>((const float*)d_in[0], inA);
    encoder(10, f1);                                    // f_rgb
    t = NB * HIN * WIN;
    prep_dep_kernel<<<(t + 255) / 256, 256, 0, stream>>>((const float*)d_in[1], inA);
    encoder(5, f2);                                     // f_lidar
    t = NB * HIN * WIN * 3;
    prep_rgb_kernel<<<(t + 255) / 256, 256, 0, stream>>>((const float*)d_in[0], inA);
    encoder(0, cn);                                     // cnet
  }

  // ---- correlation pyramid ------------------------------------------------
  corr_wmma_kernel<<<dim3(32, 32, NB), 32, 0, stream>>>(f1, f2, py0,
                                                        1.0f / sqrtf(384.f));
  { int t = NB * 512 * 8 * 16; pool2_kernel<<<(t+255)/256,256,0,stream>>>(py0, py1, 16, 32, t); }
  { int t = NB * 512 * 4 *  8; pool2_kernel<<<(t+255)/256,256,0,stream>>>(py1, py2,  8, 16, t); }
  { int t = NB * 512 * 2 *  4; pool2_kernel<<<(t+255)/256,256,0,stream>>>(py2, py3,  4,  8, t); }

  // ---- GRU init -----------------------------------------------------------
  { int t = NPIX * 192;
    init_net_inp_kernel<<<(t + 255) / 256, 256, 0, stream>>>(cn, netbuf, xbuf); }
  { int t = NB * 2 * HF * WF;
    init_coords_kernel<<<(t + 255) / 256, 256, 0, stream>>>(coords); }

  auto copy_ch = [&](const _Float16* s, int ss, int so, _Float16* d, int ds, int dof, int nch) {
    int t = NPIX * nch;
    copy_ch_kernel<<<(t + 255) / 256, 256, 0, stream>>>(s, ss, so, d, ds, dof, nch, t);
  };

  // ---- 16 RAFT update iterations -----------------------------------------
  for (int it = 0; it < 16; ++it) {
    { int t = NPIX * CORR_CH;
      corr_lookup_kernel<<<(t + 255) / 256, 256, 0, stream>>>(py0, py1, py2, py3,
                                                              coords, corrbuf); }
    flow_kernel<<<(NPIX + 255) / 256, 256, 0, stream>>>(coords, xbuf);

    conv(corrbuf,    16, 32, 324, 324, 16, corbuf, 16, 32, 256,   0, 1, ACT_RELU); // convc1
    conv(corbuf,     16, 32, 256, 256, 17, motin,  16, 32, 256,   0, 1, ACT_RELU); // convc2
    conv(xbuf + 318, 16, 32,   2, 320, 18, flobuf, 16, 32, 128,   0, 1, ACT_RELU); // convf1
    conv(flobuf,     16, 32, 128, 128, 19, motin,  16, 32, 256, 192, 1, ACT_RELU); // convf2
    conv(motin,      16, 32, 256, 256, 15, xbuf,   16, 32, 320, 192, 1, ACT_RELU); // conv -> mot

    for (int pass = 0; pass < 2; ++pass) {
      int zi = pass ? 25 : 24, ri = pass ? 23 : 22, qi = pass ? 21 : 20;
      copy_ch(netbuf, 192, 0, hxbuf, 512, 0, 192);
      copy_ch(xbuf,   320, 0, hxbuf, 512, 192, 320);
      conv(hxbuf,  16, 32, 512, 512, zi, zbuf, 16, 32, 192, 0, 1, ACT_SIG);
      conv(hxbuf,  16, 32, 512, 512, ri, rbuf, 16, 32, 192, 0, 1, ACT_SIG);
      { int t = NPIX * 192;
        rnet_kernel<<<(t + 255) / 256, 256, 0, stream>>>(netbuf, rbuf, qinbuf, t); }
      copy_ch(xbuf, 320, 0, qinbuf, 512, 192, 320);
      conv(qinbuf, 16, 32, 512, 512, qi, qbuf, 16, 32, 192, 0, 1, ACT_TANH);
      { int t = NPIX * 192;
        gru_update_kernel<<<(t + 255) / 256, 256, 0, stream>>>(netbuf, zbuf, qbuf, t); }
    }

    conv(netbuf, 16, 32, 192, 192, 26, fhbuf, 16, 32, 256, 0, 1, ACT_RELU);  // fh1
    conv(fhbuf,  16, 32, 256, 256, 27, dflow, 16, 32,   2, 0, 1, ACT_NONE);  // fh2
    { int t = NB * 2 * HF * WF;
      add_dflow_kernel<<<(t + 255) / 256, 256, 0, stream>>>(coords, dflow); }
  }

  // ---- FC head -> (t[8,3], r[8,4]) flat into d_out ------------------------
  fc_head_kernel<<<NB, 256, 0, stream>>>(coords,
      (const float*)d_in[33], (const float*)d_in[32],   // fc1
      (const float*)d_in[37], (const float*)d_in[36],   // fc1_trasl
      (const float*)d_in[35], (const float*)d_in[34],   // fc1_rot
      (const float*)d_in[41], (const float*)d_in[40],   // fc2_trasl
      (const float*)d_in[39], (const float*)d_in[38],   // fc2_rot
      (float*)d_out);
}
